// OptimizedJointGraphModel_20392504721606
// MI455X (gfx1250) — matrix operand order repaired and checked by reading, not compiled, not run on hardware
//
#include <hip/hip_runtime.h>
#include <hip/hip_bf16.h>

// ---------------------------------------------------------------------------
// Types for CDNA5 WMMA (wave32, 16x16x32 bf16 -> f32 accumulate)
// ---------------------------------------------------------------------------
typedef __attribute__((ext_vector_type(16))) __bf16 v16bf;
typedef __attribute__((ext_vector_type(8)))  float  v8f;

__device__ __forceinline__ unsigned short f2bf(float x) {
  unsigned int u = __float_as_uint(x);
  u += 0x7FFFu + ((u >> 16) & 1u);     // round-to-nearest-even
  return (unsigned short)(u >> 16);
}

__device__ __forceinline__ void atomicMaxF(float* addr, float v) {
  // IEEE-ordering trick: signed max for non-negatives, unsigned min for negatives
  if (v >= 0.f) atomicMax((int*)addr, __float_as_int(v));
  else          atomicMin((unsigned int*)addr, __float_as_uint(v));
}

// ---------------------------------------------------------------------------
// Degree / normalization
// ---------------------------------------------------------------------------
__global__ void k_deg_count(const int* __restrict__ ei, float* __restrict__ deg, int E) {
  int t = blockIdx.x * blockDim.x + threadIdx.x;
  if (t >= E) return;
  atomicAdd(&deg[ei[E + t]], 1.0f);
}

__global__ void k_deg_to_dinv(float* __restrict__ deg, int N) {
  int t = blockIdx.x * blockDim.x + threadIdx.x;
  if (t >= N) return;
  float d = deg[t] + 1.0f;                       // self-loop
  deg[t] = rsqrtf(fmaxf(d, 1e-12f));
}

// ---------------------------------------------------------------------------
// GCN1 dense part: lin = x[N,8] @ w1[8,256]  (K=8: VALU kernel, LDS broadcast)
// ---------------------------------------------------------------------------
__global__ void k_gcn1_lin(const float* __restrict__ x, const float* __restrict__ w1,
                           float* __restrict__ lin) {
  const int n = blockIdx.x;
  const int f = threadIdx.x;            // 256 threads
  __shared__ float xs[8];
  if (f < 8) xs[f] = x[n * 8 + f];
  __syncthreads();
  float s = 0.f;
#pragma unroll
  for (int k = 0; k < 8; ++k) s += xs[k] * w1[k * 256 + f];
  lin[(long)n * 256 + f] = s;
}

// ---------------------------------------------------------------------------
// GCN scatter: agg[dst] += dinv[src]*dinv[dst] * lin[src]   (32 threads/edge)
// ---------------------------------------------------------------------------
__global__ void k_gcn_spmm(const int* __restrict__ ei, const float* __restrict__ dinv,
                           const float* __restrict__ lin, float* __restrict__ agg, int E) {
  int t = blockIdx.x * blockDim.x + threadIdx.x;
  if (t >= E * 32) return;
  const int e = t >> 5, part = t & 31;
  const int s = ei[e], d = ei[E + e];
  const float nrm = dinv[s] * dinv[d];
  const float* sp = lin + (long)s * 256 + part * 8;
  float*       dp = agg + (long)d * 256 + part * 8;
#pragma unroll
  for (int i = 0; i < 8; ++i) atomicAdd(&dp[i], nrm * sp[i]);
}

// finalize: h = relu(agg + dinv^2 * lin + b)  -> bf16 (for the next WMMA GEMM)
__global__ void k_gcn_finalize(const float* __restrict__ agg, const float* __restrict__ lin,
                               const float* __restrict__ dinv, const float* __restrict__ bias,
                               unsigned short* __restrict__ hbf, int N) {
  int t = blockIdx.x * blockDim.x + threadIdx.x;
  if (t >= N * 256) return;
  const int n = t >> 8, f = t & 255;
  const float di = dinv[n];
  float v = agg[t] + di * di * lin[t] + bias[f];
  hbf[t] = f2bf(fmaxf(v, 0.f));
}

// Convert f32 weight [K=256, N=256] (row-major) -> bf16 TRANSPOSED Wt[n][k].
// Coalesced reads; transposed writes land in L2 (tiny matrix).
__global__ void k_w_to_bf16_t(const float* __restrict__ src, unsigned short* __restrict__ dst) {
  int t = blockIdx.x * blockDim.x + threadIdx.x;   // 65536 threads
  const int k = t >> 8, n = t & 255;
  dst[n * 256 + k] = f2bf(src[t]);
}

// ---------------------------------------------------------------------------
// WMMA GEMM:  Out[rows,256] = A_bf16[rows,256] @ W[256,256]  (f32 out)
// W is supplied bf16-transposed (Wt[n][k]) so every LDS fragment read is a
// contiguous ds_load_b32.
// 64 threads = 2 waves; each wave owns TWO 16-row M-tiles (rows wave*32..+31),
// so each B fragment feeds two WMMAs. LDS: 64x256 A (32KB) + 256x32 Wt (16KB).
// ---------------------------------------------------------------------------
__global__ __launch_bounds__(64) void k_gemm_bf16_n256(
    const unsigned short* __restrict__ Abf,   // [rows_pad, 256] bf16 row-major
    const unsigned short* __restrict__ Wtbf,  // [256, 256] bf16, TRANSPOSED [n][k]
    float* __restrict__ Out)                  // [rows_pad, 256] f32
{
  __shared__ __align__(16) unsigned short Alds[64 * 256];   // 32 KB
  __shared__ __align__(16) unsigned short Wtlds[256 * 32];  // 16 KB (n-major, 32 k)
  const int tid  = threadIdx.x;
  const int lane = tid & 31;
  const int wave = tid >> 5;            // 0..1
  const int hl   = lane >> 4;           // half: 0 = lanes 0-15, 1 = lanes 16-31
  const int l16  = lane & 15;
  const long mbase = (long)blockIdx.x * 64;

  { // stage A rows mbase..mbase+31 once: 1024 uint4 / 64 threads
    const uint4* s = (const uint4*)(Abf + mbase * 256);
    uint4* d = (uint4*)Alds;
#pragma unroll
    for (int i = 0; i < 16; ++i) d[tid + i * 64] = s[tid + i * 64];
  }

  v8f acc[32];                          // [2 m-tiles][16 n-tiles]
  const v8f zz = {};
#pragma unroll
  for (int i = 0; i < 32; ++i) acc[i] = zz;

  const int arow0 = wave * 32 + l16;    // local A row, m-tile 0
  const int arow1 = arow0 + 16;         // local A row, m-tile 1

  for (int ks = 0; ks < 8; ++ks) {
    __syncthreads();                    // also covers initial A staging
    { // stage Wt k-slab: Wt[n][ks*32 .. +31] for all n: 4 rows x 4 uint4 per thread
#pragma unroll
      for (int r = 0; r < 4; ++r) {
        const int n = tid * 4 + r;
        const uint4* s = (const uint4*)(Wtbf + (long)n * 256 + ks * 32);
        uint4* d = (uint4*)(Wtlds + n * 32);
#pragma unroll
        for (int i = 0; i < 4; ++i) d[i] = s[i];
      }
    }
    __syncthreads();

    // A fragments: 16-bit A 16x32 layout; pair p -> k = (p/4)*16 + hl*8 + (p%4)*2
    union { unsigned int u[8]; v16bf v; } af0, af1;
#pragma unroll
    for (int p = 0; p < 8; ++p) {
      const int kloc = ((p >> 2) << 4) + (hl << 3) + ((p & 3) << 1);
      af0.u[p] = *(const unsigned int*)&Alds[arow0 * 256 + ks * 32 + kloc];
      af1.u[p] = *(const unsigned int*)&Alds[arow1 * 256 + ks * 32 + kloc];
    }

#pragma unroll
    for (int nt = 0; nt < 16; ++nt) {
      // B fragment: pair p -> k = (p/4)*16 + hl*8 + (p%4)*2, col = l16; pairs are contiguous
      // dwords in the transposed slab -> 8x ds_load_b32, no packing ALU.
      union { unsigned int u[8]; v16bf v; } bfr;
      const int ncol = nt * 16 + l16;
#pragma unroll
      for (int p = 0; p < 8; ++p)
        bfr.u[p] = *(const unsigned int*)&Wtlds[ncol * 32 + ((p >> 2) << 4) + (hl << 3) + ((p & 3) << 1)];

      acc[nt]      = __builtin_amdgcn_wmma_f32_16x16x32_bf16(
          false, af0.v, false, bfr.v, (short)0, acc[nt], false, false);
      acc[16 + nt] = __builtin_amdgcn_wmma_f32_16x16x32_bf16(
          false, af1.v, false, bfr.v, (short)0, acc[16 + nt], false, false);
    }
  }

  // C/D layout: VGPR r -> row = r + 8*hl, col = l16
#pragma unroll
  for (int mt = 0; mt < 2; ++mt) {
#pragma unroll
    for (int nt = 0; nt < 16; ++nt) {
#pragma unroll
      for (int r = 0; r < 8; ++r) {
        const long grow = mbase + wave * 32 + mt * 16 + hl * 8 + r;
        Out[grow * 256 + nt * 16 + l16] = acc[mt * 16 + nt][r];
      }
    }
  }
}

// ---------------------------------------------------------------------------
// GAT: attention scores, softmax (3 edge passes over E+N edges incl. self)
// ---------------------------------------------------------------------------
__global__ void k_gat_scores(const float* __restrict__ hG, const float* __restrict__ att_src,
                             const float* __restrict__ att_dst,
                             float* __restrict__ a_s, float* __restrict__ a_d, int N) {
  int t = blockIdx.x * blockDim.x + threadIdx.x;
  if (t >= N * 4) return;
  const int n = t >> 2, h = t & 3;
  const float* hp = hG + (long)n * 256 + h * 64;
  float ss = 0.f, sd = 0.f;
  for (int c = 0; c < 64; ++c) { float v = hp[c]; ss += v * att_src[h * 64 + c]; sd += v * att_dst[h * 64 + c]; }
  a_s[t] = ss; a_d[t] = sd;
}

__global__ void k_fill(float* __restrict__ p, float v, int n) {
  int t = blockIdx.x * blockDim.x + threadIdx.x;
  if (t < n) p[t] = v;
}

__device__ __forceinline__ void edge_sd(const int* ei, int e, int E, int& s, int& d) {
  if (e < E) { s = ei[e]; d = ei[E + e]; } else { s = d = e - E; }
}

__global__ void k_gat_max(const int* __restrict__ ei, const float* __restrict__ a_s,
                          const float* __restrict__ a_d, float* __restrict__ mx, int E, int N) {
  int t = blockIdx.x * blockDim.x + threadIdx.x;
  const int EE = E + N;
  if (t >= EE * 4) return;
  const int e = t >> 2, h = t & 3;
  int s, d; edge_sd(ei, e, E, s, d);
  float v = a_s[s * 4 + h] + a_d[d * 4 + h];
  v = v > 0.f ? v : 0.2f * v;
  atomicMaxF(&mx[d * 4 + h], v);
}

__global__ void k_gat_sum(const int* __restrict__ ei, const float* __restrict__ a_s,
                          const float* __restrict__ a_d, const float* __restrict__ mx,
                          float* __restrict__ denom, int E, int N) {
  int t = blockIdx.x * blockDim.x + threadIdx.x;
  const int EE = E + N;
  if (t >= EE * 4) return;
  const int e = t >> 2, h = t & 3;
  int s, d; edge_sd(ei, e, E, s, d);
  float v = a_s[s * 4 + h] + a_d[d * 4 + h];
  v = v > 0.f ? v : 0.2f * v;
  atomicAdd(&denom[d * 4 + h], __expf(v - mx[d * 4 + h]));
}

__global__ void k_gat_scatter(const int* __restrict__ ei, const float* __restrict__ a_s,
                              const float* __restrict__ a_d, const float* __restrict__ mx,
                              const float* __restrict__ denom, const float* __restrict__ hG,
                              float* __restrict__ outb, int E, int N) {
  int t = blockIdx.x * blockDim.x + threadIdx.x;
  const int EE = E + N;
  if (t >= EE * 16) return;
  const int e = t >> 4, q = t & 15;
  const int h = q >> 2, chunk = q & 3;
  int s, d; edge_sd(ei, e, E, s, d);
  float v = a_s[s * 4 + h] + a_d[d * 4 + h];
  v = v > 0.f ? v : 0.2f * v;
  const float coef = __expf(v - mx[d * 4 + h]) / denom[d * 4 + h];
  const float* sp = hG   + (long)s * 256 + h * 64 + chunk * 16;
  float*       dp = outb + (long)d * 256 + h * 64 + chunk * 16;
#pragma unroll
  for (int i = 0; i < 16; ++i) atomicAdd(&dp[i], coef * sp[i]);
}

__global__ void k_gat_finalize(float* __restrict__ hio, const float* __restrict__ bias, int N) {
  int t = blockIdx.x * blockDim.x + threadIdx.x;
  if (t >= N * 256) return;
  hio[t] = fmaxf(hio[t] + bias[t & 255], 0.f);
}

// ---------------------------------------------------------------------------
// Mean pool + fused MLP head
// ---------------------------------------------------------------------------
__global__ void k_pool(const float* __restrict__ h, const int* __restrict__ batch,
                       float* __restrict__ sums, float* __restrict__ cnt, int N) {
  int t = blockIdx.x * blockDim.x + threadIdx.x;
  if (t >= N * 256) return;
  const int n = t >> 8, f = t & 255;
  const int g = batch[n];
  atomicAdd(&sums[g * 256 + f], h[t]);
  if (f == 0) atomicAdd(&cnt[g], 1.f);
}

__global__ __launch_bounds__(256) void k_head(
    const float* __restrict__ sums, const float* __restrict__ cnt,
    const float* __restrict__ desc,
    const float* dw1, const float* db1, const float* dw2, const float* db2,
    const float* fw1, const float* fb1, const float* fw2, const float* fb2,
    const float* pw1, const float* pb1, const float* pw2, const float* pb2,
    const float* pw3, const float* pb3, float* __restrict__ out) {
  const int g = blockIdx.x, t = threadIdx.x;
  __shared__ float comb[384];
  __shared__ float b1[256];
  __shared__ float b2[128];
  __shared__ float t1[64];
  __shared__ float p1[64];
  __shared__ float p2[64];

  const float c = fmaxf(cnt[g], 1.0f);
  if (t < 256) comb[t] = sums[g * 256 + t] / c;
  if (t < 64) {
    float s = db1[t];
    for (int k = 0; k < 8; ++k) s += desc[g * 8 + k] * dw1[k * 64 + t];
    t1[t] = fmaxf(s, 0.f);
  }
  __syncthreads();
  if (t < 128) {
    float s = db2[t];
    for (int k = 0; k < 64; ++k) s += t1[k] * dw2[k * 128 + t];
    comb[256 + t] = s;
  }
  __syncthreads();
  if (t < 256) {
    float s = fb1[t];
    for (int k = 0; k < 384; ++k) s += comb[k] * fw1[k * 256 + t];
    b1[t] = fmaxf(s, 0.f);
  }
  __syncthreads();
  if (t < 128) {
    float s = fb2[t];
    for (int k = 0; k < 256; ++k) s += b1[k] * fw2[k * 128 + t];
    b2[t] = s;
  }
  __syncthreads();
  if (t < 64) {
    float s = pb1[t];
    for (int k = 0; k < 128; ++k) s += b2[k] * pw1[k * 64 + t];
    p1[t] = fmaxf(s, 0.f);
  }
  __syncthreads();
  if (t < 64) {
    float s = pb2[t];
    for (int k = 0; k < 64; ++k) s += p1[k] * pw2[k * 64 + t];
    p2[t] = fmaxf(s, 0.f);
  }
  __syncthreads();
  if (t == 0) {
    float s = pb3[0];
    for (int k = 0; k < 64; ++k) s += p2[k] * pw3[k];
    out[g] = 1.f / (1.f + __expf(-s));
  }
}

// ---------------------------------------------------------------------------
// Launch
// ---------------------------------------------------------------------------
extern "C" void kernel_launch(void* const* d_in, const int* in_sizes, int n_in,
                              void* d_out, int out_size, void* d_ws, size_t ws_size,
                              hipStream_t stream) {
  const float* x       = (const float*)d_in[0];
  const float* desc    = (const float*)d_in[1];
  const int*   ei      = (const int*)  d_in[2];
  const int*   batch   = (const int*)  d_in[3];
  const float* gcn1_w  = (const float*)d_in[4];
  const float* gcn1_b  = (const float*)d_in[5];
  const float* gcn2_w  = (const float*)d_in[6];
  const float* gcn2_b  = (const float*)d_in[7];
  const float* gat_w   = (const float*)d_in[8];
  const float* att_src = (const float*)d_in[9];
  const float* att_dst = (const float*)d_in[10];
  const float* gat_b   = (const float*)d_in[11];
  const float* dw1 = (const float*)d_in[12]; const float* db1 = (const float*)d_in[13];
  const float* dw2 = (const float*)d_in[14]; const float* db2 = (const float*)d_in[15];
  const float* fw1 = (const float*)d_in[16]; const float* fb1 = (const float*)d_in[17];
  const float* fw2 = (const float*)d_in[18]; const float* fb2 = (const float*)d_in[19];
  const float* pw1 = (const float*)d_in[20]; const float* pb1 = (const float*)d_in[21];
  const float* pw2 = (const float*)d_in[22]; const float* pb2 = (const float*)d_in[23];
  const float* pw3 = (const float*)d_in[24]; const float* pb3 = (const float*)d_in[25];

  const int N = in_sizes[0] / 8;
  const int E = in_sizes[2] / 2;
  const int G = in_sizes[1] / 8;
  const int NPAD = ((N + 63) / 64) * 64;

  // Workspace carve-out (256B aligned slabs)
  char* wp = (char*)d_ws;
  auto carve = [&](size_t bytes) -> char* {
    char* p = wp; wp += (bytes + 255) & ~(size_t)255; return p;
  };
  float*          lin   = (float*)         carve((size_t)NPAD * 256 * 4); // lin1 / lin2 / hG
  float*          agg   = (float*)         carve((size_t)NPAD * 256 * 4); // aggregation / GAT out
  unsigned short* hbf   = (unsigned short*)carve((size_t)NPAD * 256 * 2); // bf16 activations
  unsigned short* w2bf  = (unsigned short*)carve((size_t)256 * 256 * 2);  // bf16 Wt (transposed)
  unsigned short* gwbf  = (unsigned short*)carve((size_t)256 * 256 * 2);  // bf16 Wt (transposed)
  float*          dinv  = (float*)         carve((size_t)N * 4);
  float*          a_s   = (float*)         carve((size_t)N * 4 * 4);
  float*          a_d   = (float*)         carve((size_t)N * 4 * 4);
  float*          mx    = (float*)         carve((size_t)N * 4 * 4);
  float*          denom = (float*)         carve((size_t)N * 4 * 4);
  float*          sums  = (float*)         carve((size_t)G * 256 * 4);
  float*          cnt   = (float*)         carve((size_t)G * 4);

  const int TB = 256;
  auto nb = [](long n, int b) { return (int)((n + b - 1) / b); };
  const long EE = (long)E + N;

  // --- GCN layer 1 -------------------------------------------------------
  hipMemsetAsync(dinv, 0, (size_t)N * 4, stream);
  k_deg_count  <<<nb(E, TB), TB, 0, stream>>>(ei, dinv, E);
  k_deg_to_dinv<<<nb(N, TB), TB, 0, stream>>>(dinv, N);
  k_gcn1_lin   <<<N, 256, 0, stream>>>(x, gcn1_w, lin);
  hipMemsetAsync(agg, 0, (size_t)N * 256 * 4, stream);
  k_gcn_spmm    <<<nb((long)E * 32, TB), TB, 0, stream>>>(ei, dinv, lin, agg, E);
  k_gcn_finalize<<<nb((long)N * 256, TB), TB, 0, stream>>>(agg, lin, dinv, gcn1_b, hbf, N);

  // --- GCN layer 2 (WMMA GEMM) ------------------------------------------
  k_w_to_bf16_t  <<<nb(256 * 256, TB), TB, 0, stream>>>(gcn2_w, w2bf);
  k_gemm_bf16_n256<<<NPAD / 64, 64, 0, stream>>>(hbf, w2bf, lin);
  hipMemsetAsync(agg, 0, (size_t)N * 256 * 4, stream);
  k_gcn_spmm    <<<nb((long)E * 32, TB), TB, 0, stream>>>(ei, dinv, lin, agg, E);
  k_gcn_finalize<<<nb((long)N * 256, TB), TB, 0, stream>>>(agg, lin, dinv, gcn2_b, hbf, N);

  // --- GAT layer (WMMA GEMM + edge softmax) ------------------------------
  k_w_to_bf16_t  <<<nb(256 * 256, TB), TB, 0, stream>>>(gat_w, gwbf);
  k_gemm_bf16_n256<<<NPAD / 64, 64, 0, stream>>>(hbf, gwbf, lin);   // lin = hG
  k_gat_scores <<<nb((long)N * 4, TB), TB, 0, stream>>>(lin, att_src, att_dst, a_s, a_d, N);
  k_fill       <<<nb((long)N * 4, TB), TB, 0, stream>>>(mx, -1e30f, N * 4);
  k_gat_max    <<<nb(EE * 4, TB), TB, 0, stream>>>(ei, a_s, a_d, mx, E, N);
  hipMemsetAsync(denom, 0, (size_t)N * 4 * 4, stream);
  k_gat_sum    <<<nb(EE * 4, TB), TB, 0, stream>>>(ei, a_s, a_d, mx, denom, E, N);
  hipMemsetAsync(agg, 0, (size_t)N * 256 * 4, stream);
  k_gat_scatter<<<nb(EE * 16, TB), TB, 0, stream>>>(ei, a_s, a_d, mx, denom, lin, agg, E, N);
  k_gat_finalize<<<nb((long)N * 256, TB), TB, 0, stream>>>(agg, gat_b, N);

  // --- Pool + head -------------------------------------------------------
  hipMemsetAsync(sums, 0, (size_t)G * 256 * 4, stream);
  hipMemsetAsync(cnt, 0, (size_t)G * 4, stream);
  k_pool<<<nb((long)N * 256, TB), TB, 0, stream>>>(agg, batch, sums, cnt, N);
  k_head<<<G, 256, 0, stream>>>(sums, cnt, desc,
                                dw1, db1, dw2, db2, fw1, fb1, fw2, fb2,
                                pw1, pb1, pw2, pb2, pw3, pb3, (float*)d_out);
}
